// MHSA_22797686407456
// MI455X (gfx1250) — matrix operand order repaired and checked by reading, not compiled, not run on hardware
//
#include <hip/hip_runtime.h>
#include <math.h>

// ---------------------------------------------------------------------------
// MHSA for (B=32, C=512, 32x32), HEADS=8, D=64, N=1024 on gfx1250 (CDNA5).
// f16 WMMA (v_wmma_f32_16x16x32_f16) everywhere, flash-attention streaming
// softmax, and ASYNCcnt-tracked global->LDS staging (global_load_async_to_lds)
// so each block loads shared K/Q/V fragments once instead of once per wave.
// ---------------------------------------------------------------------------

typedef __attribute__((ext_vector_type(16))) _Float16 v16h;
typedef __attribute__((ext_vector_type(8)))  _Float16 v8h;
typedef __attribute__((ext_vector_type(8)))  float    v8f;
typedef __attribute__((ext_vector_type(4)))  float    f32x4;

#define BATCH 32
#define CCH   512
#define HEADS 8
#define DH    64
#define NPIX  1024

static __device__ __forceinline__ v8f wmma_f16(v16h a, v16h b, v8f c) {
    return __builtin_amdgcn_wmma_f32_16x16x32_f16(
        /*neg_a=*/false, a, /*neg_b=*/false, b,
        /*c_mod=*/(short)0, c, /*reuse_a=*/false, /*reuse_b=*/false);
}

// LDS byte address (within the LDS aperture) of a generic pointer to __shared__.
static __device__ __forceinline__ unsigned lds_off(const void* p) {
    return (unsigned)(unsigned long long)(__attribute__((address_space(3))) const void*)p;
}

// Async global->LDS 16-byte copy, tracked by ASYNCcnt (CDNA5).
static __device__ __forceinline__ void async_copy_b128(unsigned ldsAddr, const void* g) {
    asm volatile("global_load_async_to_lds_b128 %0, %1, off"
                 :: "v"(ldsAddr), "v"((unsigned long long)g)
                 : "memory");
}

static __device__ __forceinline__ void wait_async_all() {
    asm volatile("s_wait_asynccnt 0x0" ::: "memory");
}

// ---------------------------------------------------------------------------
// Phase 0: posT[h][n][d] = rel_h[h,d,y] + rel_w[h,d,w], n = w*32 + y  (f16)
// ---------------------------------------------------------------------------
__global__ void pos_prep_kernel(const float* __restrict__ rel_h,
                                const float* __restrict__ rel_w,
                                _Float16* __restrict__ posT) {
    int idx = blockIdx.x * blockDim.x + threadIdx.x;
    if (idx >= HEADS * NPIX * DH) return;
    int d = idx & 63;
    int n = (idx >> 6) & 1023;
    int h = idx >> 16;
    int w = n >> 5;
    int y = n & 31;
    float v = rel_h[(h * DH + d) * 32 + y] + rel_w[(h * DH + d) * 32 + w];
    posT[idx] = (_Float16)v;
}

// ---------------------------------------------------------------------------
// Phase 1: projections. Y[o,n] = sum_c W[o,c] * X[b,c,n] + bias[o].
// One wave computes one 16x16 tile (M=o, N=n), K=512 in 16 wmma steps.
// p==0 -> qpT [b][h][n][d] f16 ; p==1 -> kpT [b][h][n][d] ; p==2 -> vp [b][o][n]
// grid: (NPIX/64, CCH/16, 3*BATCH), block 128 (4 waves, n-subtiles)
// ---------------------------------------------------------------------------
__global__ __launch_bounds__(128) void proj_gemm_kernel(
    const float* __restrict__ qi, const float* __restrict__ ki, const float* __restrict__ vi,
    const float* __restrict__ Wq, const float* __restrict__ bq,
    const float* __restrict__ Wk, const float* __restrict__ bk,
    const float* __restrict__ Wv, const float* __restrict__ bv,
    _Float16* __restrict__ qpT, _Float16* __restrict__ kpT, _Float16* __restrict__ vp)
{
    const int lane = threadIdx.x & 31;
    const int wave = threadIdx.x >> 5;
    const int b    = blockIdx.z & 31;
    const int p    = blockIdx.z >> 5;           // 0=q 1=k 2=v
    const int o0   = blockIdx.y * 16;
    const int n0   = blockIdx.x * 64 + wave * 16;

    const float* X;  const float* W;  const float* bias;
    if (p == 0)      { X = qi; W = Wq; bias = bq; }
    else if (p == 1) { X = ki; W = Wk; bias = bk; }
    else             { X = vi; W = Wv; bias = bv; }
    X += (size_t)b * CCH * NPIX;

    const int col = lane & 15;        // A row index == B/D column index
    const int g   = lane >> 4;        // half-wave group
    const int hiA = g * 8;            // A K-offset per CDNA5 16-bit A layout
    const int hiB = g * 16;           // contiguous-order K-offset (unused here; B uses hiA)

    const float* wrow = W + (size_t)(o0 + col) * CCH;
    const int    ncol = n0 + col;

    v8f acc = {};
    #pragma unroll 4
    for (int k0 = 0; k0 < CCH; k0 += 32) {
        __builtin_prefetch(wrow + k0 + 64, 0, 0);
        // ---- A fragment: 16 f16 from W row (two contiguous 8-float chunks)
        f32x4 a0 = *(const f32x4*)(wrow + k0 + hiA);
        f32x4 a1 = *(const f32x4*)(wrow + k0 + hiA + 4);
        f32x4 a2 = *(const f32x4*)(wrow + k0 + hiA + 16);
        f32x4 a3 = *(const f32x4*)(wrow + k0 + hiA + 20);
        v16h A;
        #pragma unroll
        for (int i = 0; i < 4; ++i) {
            A[i]      = (_Float16)a0[i];
            A[4 + i]  = (_Float16)a1[i];
            A[8 + i]  = (_Float16)a2[i];
            A[12 + i] = (_Float16)a3[i];
        }
        // ---- B fragment: 16 f16 from X column (stride NPIX; coalesced over lanes)
        const float* xcol = X + (size_t)(k0 + hiA) * NPIX + ncol;
        v16h Bf;
        #pragma unroll
        for (int i = 0; i < 16; ++i) Bf[i] = (_Float16)xcol[(size_t)(i + (i & 8)) * NPIX];

        acc = wmma_f16(A, Bf, acc);
    }

    // bias for rows o = o0 + 8g + r
    f32x4 bb0 = *(const f32x4*)(bias + o0 + 8 * g);
    f32x4 bb1 = *(const f32x4*)(bias + o0 + 8 * g + 4);
    float bs[8];
    #pragma unroll
    for (int i = 0; i < 4; ++i) { bs[i] = bb0[i]; bs[4 + i] = bb1[i]; }

    if (p < 2) {
        // store transposed f16: dst[b][h][n][d], contiguous over r (d-dim)
        _Float16* dst  = (p == 0) ? qpT : kpT;
        const int head = o0 >> 6;
        const int d0   = (o0 & 63) + 8 * g;
        _Float16* ptr  = dst + (((size_t)(b * HEADS + head) * NPIX) + ncol) * DH + d0;
        v8h ph;
        #pragma unroll
        for (int r = 0; r < 8; ++r) ph[r] = (_Float16)(acc[r] + bs[r]);
        *(v8h*)ptr = ph;
    } else {
        // store vp[b][o][n] f16 (n-contiguous rows; strided per-lane scalar stores)
        #pragma unroll
        for (int r = 0; r < 8; ++r) {
            vp[((size_t)b * CCH + o0 + 8 * g + r) * NPIX + ncol] =
                (_Float16)(acc[r] + bs[r]);
        }
    }
}

// ---------------------------------------------------------------------------
// Phase 2: flash attention. 4 waves/block, one 16-query tile each, same (b,h).
// Per 32-key chunk the block async-stages the *shared* operands into LDS once:
//   ldsK = kpT rows m0..m0+31   (4KB, contiguous slab)
//   ldsQ = qpT rows m0..m0+31   (4KB, contiguous slab)   [aug K 64..127]
//   ldsV = vp[d=0..63][m0..+31] (4KB, 64 rows x 64B)
// then S(16x32) = [Qp;Pos]^T [Kp;Qp] (8 wmma), online softmax, P repack via
// per-wave LDS tile, O(16x64) += P x V^T (4 wmma).
// grid: (NPIX/64, B*HEADS), block 128
// ---------------------------------------------------------------------------
__global__ __launch_bounds__(128) void attn_kernel(
    const _Float16* __restrict__ qpT, const _Float16* __restrict__ kpT,
    const _Float16* __restrict__ vp,  const _Float16* __restrict__ posT,
    float* __restrict__ out)
{
    __shared__ _Float16 smem[8192];          // 16KB
    _Float16* ldsK = smem;                   // 2048 halfs
    _Float16* ldsQ = smem + 2048;            // 2048 halfs
    _Float16* ldsV = smem + 4096;            // 2048 halfs
    _Float16* ldsP = smem + 6144;            // 4 waves x 512 halfs

    const int tid  = threadIdx.x;
    const int lane = tid & 31;
    const int wave = tid >> 5;
    const int bh   = blockIdx.y;
    const int b    = bh >> 3;
    const int h    = bh & 7;
    const int n0   = (blockIdx.x * 4 + wave) * 16;

    const int col = lane & 15;
    const int g   = lane >> 4;
    const int hiA = g * 8;
    const int hiB = g * 16;

    _Float16* Pl = ldsP + wave * (16 * 32);

    const unsigned kOff = lds_off(ldsK);     // LDS byte addresses for async DMA
    const unsigned qOff = lds_off(ldsQ);
    const unsigned vOff = lds_off(ldsV);

    // ---- loop-invariant A fragments for the augmented query matrix (K=128)
    const _Float16* qrow = qpT + ((size_t)(b * HEADS + h) * NPIX + (n0 + col)) * DH;
    const _Float16* prow = posT + ((size_t)h * NPIX + (n0 + col)) * DH;
    v16h Afrag[4];
    #pragma unroll
    for (int ks = 0; ks < 2; ++ks) {
        v8h lo = *(const v8h*)(qrow + ks * 32 + hiB);
        v8h hi = *(const v8h*)(qrow + ks * 32 + hiB + 8);
        Afrag[ks] = __builtin_shufflevector(lo, hi, 0,1,2,3,4,5,6,7,8,9,10,11,12,13,14,15);
    }
    #pragma unroll
    for (int ks = 0; ks < 2; ++ks) {
        v8h lo = *(const v8h*)(prow + ks * 32 + hiB);
        v8h hi = *(const v8h*)(prow + ks * 32 + hiB + 8);
        Afrag[2 + ks] = __builtin_shufflevector(lo, hi, 0,1,2,3,4,5,6,7,8,9,10,11,12,13,14,15);
    }

    const char* gK = (const char*)(kpT + (size_t)(b * HEADS + h) * NPIX * DH);
    const char* gQ = (const char*)(qpT + (size_t)(b * HEADS + h) * NPIX * DH);
    const char* gV = (const char*)(vp + ((size_t)b * CCH + h * DH) * NPIX);

    v8f  Oacc[4] = {{}, {}, {}, {}};
    float mrow[8], lrow[8];
    #pragma unroll
    for (int r = 0; r < 8; ++r) { mrow[r] = -1.0e30f; lrow[r] = 0.0f; }

    for (int m0 = 0; m0 < NPIX; m0 += 32) {
        // ---- async-stage shared K/Q/V operands for this key chunk (12KB/block)
        {
            const char* srcK = gK + (size_t)m0 * DH * 2;   // contiguous 4KB slab
            const char* srcQ = gQ + (size_t)m0 * DH * 2;   // contiguous 4KB slab
            #pragma unroll
            for (int j = 0; j < 2; ++j) {
                const unsigned lin = (unsigned)tid * 16 + j * 2048;
                async_copy_b128(kOff + lin, srcK + lin);
                async_copy_b128(qOff + lin, srcQ + lin);
                const unsigned d  = lin >> 6;              // V: row d, 64B per row
                const unsigned of = lin & 63;
                async_copy_b128(vOff + lin, gV + (size_t)d * (NPIX * 2) + m0 * 2 + of);
            }
            wait_async_all();
            __syncthreads();
        }

        // ---- S = Aaug^T Baug for this 16x32 key chunk (B frags from LDS)
        v8f S0 = {}, S1 = {};
        #pragma unroll
        for (int ks = 0; ks < 4; ++ks) {
            const _Float16* base = (ks < 2) ? ldsK : ldsQ;
            const int koff = (ks & 1) * 32 + hiB;
            v16h B0 = *(const v16h*)(base + (col) * DH + koff);
            v16h B1 = *(const v16h*)(base + (16 + col) * DH + koff);
            S0 = wmma_f16(Afrag[ks], B0, S0);
            S1 = wmma_f16(Afrag[ks], B1, S1);
        }

        // ---- online softmax (rows live on VGPR r across 16-lane groups)
        float scale[8], p0[8], p1[8];
        #pragma unroll
        for (int r = 0; r < 8; ++r) {
            float cm = fmaxf(S0[r], S1[r]);
            cm = fmaxf(cm, __shfl_xor(cm, 1, 32));
            cm = fmaxf(cm, __shfl_xor(cm, 2, 32));
            cm = fmaxf(cm, __shfl_xor(cm, 4, 32));
            cm = fmaxf(cm, __shfl_xor(cm, 8, 32));
            float mn = fmaxf(mrow[r], cm);
            scale[r] = __expf(mrow[r] - mn);
            p0[r]    = __expf(S0[r] - mn);
            p1[r]    = __expf(S1[r] - mn);
            float rs = p0[r] + p1[r];
            rs += __shfl_xor(rs, 1, 32);
            rs += __shfl_xor(rs, 2, 32);
            rs += __shfl_xor(rs, 4, 32);
            rs += __shfl_xor(rs, 8, 32);
            lrow[r] = lrow[r] * scale[r] + rs;
            mrow[r] = mn;
        }
        #pragma unroll
        for (int t = 0; t < 4; ++t) {
            #pragma unroll
            for (int r = 0; r < 8; ++r) Oacc[t][r] *= scale[r];
        }

        // ---- repack P (D-layout) -> A-fragment layout through per-wave LDS
        #pragma unroll
        for (int r = 0; r < 8; ++r) {
            Pl[(8 * g + r) * 32 + col]      = (_Float16)p0[r];
            Pl[(8 * g + r) * 32 + 16 + col] = (_Float16)p1[r];
        }
        __syncthreads();
        v16h PA;
        {
            const _Float16* pp = Pl + col * 32 + hiB;
            v8h lo = *(const v8h*)pp;
            v8h hi = *(const v8h*)(pp + 8);
            PA = __builtin_shufflevector(lo, hi, 0,1,2,3,4,5,6,7,8,9,10,11,12,13,14,15);
        }

        // ---- O += P x V^T (B fragments from staged LDS, contiguous in n)
        #pragma unroll
        for (int dt = 0; dt < 4; ++dt) {
            v16h Bv = *(const v16h*)(ldsV + (dt * 16 + col) * 32 + hiB);
            Oacc[dt] = wmma_f16(PA, Bv, Oacc[dt]);
        }
        __syncthreads();   // staged buffers free for next chunk's DMA
    }

    // ---- epilogue: normalize and store f32 output (b, o=h*64+d, n)
    float inv[8];
    #pragma unroll
    for (int r = 0; r < 8; ++r) inv[r] = 1.0f / lrow[r];
    #pragma unroll
    for (int dt = 0; dt < 4; ++dt) {
        const size_t obase = ((size_t)b * CCH + h * DH + dt * 16 + col) * NPIX + n0 + 8 * g;
        #pragma unroll
        for (int r = 0; r < 8; ++r) out[obase + r] = Oacc[dt][r] * inv[r];
    }
}

// ---------------------------------------------------------------------------
extern "C" void kernel_launch(void* const* d_in, const int* in_sizes, int n_in,
                              void* d_out, int out_size, void* d_ws, size_t ws_size,
                              hipStream_t stream) {
    (void)in_sizes; (void)n_in; (void)out_size; (void)ws_size;
    const float* q     = (const float*)d_in[0];
    const float* k     = (const float*)d_in[1];
    const float* v     = (const float*)d_in[2];
    const float* Wq    = (const float*)d_in[3];
    const float* bq    = (const float*)d_in[4];
    const float* Wk    = (const float*)d_in[5];
    const float* bk    = (const float*)d_in[6];
    const float* Wv    = (const float*)d_in[7];
    const float* bv    = (const float*)d_in[8];
    const float* rel_h = (const float*)d_in[9];
    const float* rel_w = (const float*)d_in[10];

    const size_t projElems = (size_t)BATCH * CCH * NPIX;      // 16M halfs = 32MB
    _Float16* qpT  = (_Float16*)d_ws;
    _Float16* kpT  = qpT + projElems;
    _Float16* vp   = kpT + projElems;
    _Float16* posT = vp  + projElems;                          // +1MB

    // Phase 0: relative-position table, transposed to [h][n][d] f16
    {
        const int total = HEADS * NPIX * DH;
        pos_prep_kernel<<<dim3((total + 255) / 256), dim3(256), 0, stream>>>(
            rel_h, rel_w, posT);
    }
    // Phase 1: q/k/v projections -> layout-specialized f16 intermediates
    proj_gemm_kernel<<<dim3(NPIX / 64, CCH / 16, 3 * BATCH), dim3(128), 0, stream>>>(
        q, k, v, Wq, bq, Wk, bk, Wv, bv, qpT, kpT, vp);
    // Phase 2: flash attention with async-staged shared operands
    attn_kernel<<<dim3(NPIX / 64, BATCH * HEADS), dim3(128), 0, stream>>>(
        qpT, kpT, vp, posT, (float*)d_out);
}
